// GatedLatticeLayer_28836410425745
// MI455X (gfx1250) — compile-verified, matrix operand order checked by reading, not run on hardware
//
#include <hip/hip_runtime.h>
#include <hip/hip_bf16.h>
#include <math.h>

// Problem dims (fixed by the reference)
#define BDIM 16
#define SDIM 512
#define WDIM 5
#define HDIM 512
#define EDIM 256
#define VDIM 100000
#define BS   (BDIM * SDIM)   // 8192 positions

typedef __attribute__((ext_vector_type(16))) __bf16 v16bf;
typedef __attribute__((ext_vector_type(8)))  __bf16 v8bf;
typedef __attribute__((ext_vector_type(8)))  float  v8f;
typedef __attribute__((ext_vector_type(4)))  unsigned int u32x4;
typedef __attribute__((ext_vector_type(8)))  int i32x8;
typedef __attribute__((ext_vector_type(4)))  int i32x4;

// ---------------------------------------------------------------------------
// WMMA bf16 fragment load from a row-major bf16 matrix (rows = M for A, N for B,
// since all B operands are stored transposed N x K).
// Per CDNA5 ISA: lane l -> row = l&15, K-half = l>>4; element e ->
// k = khalf*8 + (e&7) + (e&8 ? 16 : 0). Each lane therefore needs two
// contiguous 8-element (16-byte) chunks at k offsets khalf*8 and 16+khalf*8.
// ---------------------------------------------------------------------------
static __device__ __forceinline__ v16bf frag_ld(const __bf16* __restrict__ base, int ld,
                                                int rbase, int kbase, int lane) {
  const __bf16* p = base + (size_t)(rbase + (lane & 15)) * ld + kbase + ((lane >> 4) & 1) * 8;
  v8bf lo = *(const v8bf*)p;          // global_load_b128
  v8bf hi = *(const v8bf*)(p + 16);   // global_load_b128
  return __builtin_shufflevector(lo, hi, 0, 1, 2, 3, 4, 5, 6, 7,
                                 8, 9, 10, 11, 12, 13, 14, 15);
}

#define WMMA_BF16(a, b, c) \
  __builtin_amdgcn_wmma_f32_16x16x32_bf16(false, (a), false, (b), (short)0, (c), false, false)

// ---------------------------------------------------------------------------
// K0: one-time f32 -> bf16 conversions / transposes of all GEMM operands.
//   hidb [8192x512]  = hiddens
//   WhTb [1536x512]  = W_hidden^T            (N x K for K1's B)
//   Wkb  [ 768x512]  : row t*256+e = key half of W_t row e   (N x K for K2's B)
//   WvTb [ 512x768]  : WvTb[n][t*256+e] = W_t[e][512+n]      (N x K for K4's B)
// ---------------------------------------------------------------------------
__global__ __launch_bounds__(256) void k0_prep(const float* __restrict__ hiddens,
                                               const float* __restrict__ Wh,
                                               const float* __restrict__ Wb,
                                               const float* __restrict__ Wm,
                                               const float* __restrict__ We,
                                               __bf16* __restrict__ hidb,
                                               __bf16* __restrict__ WhTb,
                                               __bf16* __restrict__ Wkb,
                                               __bf16* __restrict__ WvTb) {
  int i = blockIdx.x * 256 + threadIdx.x;
  const int n0 = BS * HDIM;        // 4194304
  const int n1 = 1536 * 512;       //  786432
  const int n2 = 768 * 512;        //  393216
  const int n3 = 512 * 768;        //  393216
  if (i < n0) { hidb[i] = (__bf16)hiddens[i]; return; }
  i -= n0;
  if (i < n1) {
    const int n = i / 512, k = i - n * 512;
    WhTb[i] = (__bf16)Wh[(long)k * 1536 + n];
    return;
  }
  i -= n1;
  if (i < n2) {
    const int row = i / 512, h = i - row * 512;
    const int t = row >> 8, e = row & 255;
    const float* W = (t == 0) ? Wb : (t == 1) ? Wm : We;
    Wkb[i] = (__bf16)W[(long)e * 1024 + h];
    return;
  }
  i -= n2;
  if (i < n3) {
    const int n = i / 768, kk = i - n * 768;
    const int t = kk >> 8, e = kk & 255;
    const float* W = (t == 0) ? Wb : (t == 1) ? Wm : We;
    WvTb[i] = (__bf16)W[(long)e * 1024 + 512 + n];
  }
}
#define K0_TOTAL (BS * HDIM + 1536 * 512 + 768 * 512 + 512 * 768)

// ---------------------------------------------------------------------------
// K1: qkv = hiddens @ W_hidden + b  (8192x512 @ 512x1536), also emits bf16 q.
// Block = 8 waves, tile 64(M) x 256(N); wave tile 32 x 64.
// ---------------------------------------------------------------------------
__global__ __launch_bounds__(256) void k1_qkv(const __bf16* __restrict__ hidb,
                                              const __bf16* __restrict__ WhTb,
                                              const float* __restrict__ bh,
                                              float* __restrict__ qkv,
                                              __bf16* __restrict__ qb16) {
  const int lane = threadIdx.x & 31, wid = threadIdx.x >> 5;
  const int mblk = blockIdx.y * 64 + (wid & 1) * 32;
  const int nblk = blockIdx.x * 256 + (wid >> 1) * 64;
  v8f acc[2][4] = {};
  for (int k = 0; k < HDIM; k += 32) {
    v16bf a0 = frag_ld(hidb, HDIM, mblk, k, lane);
    v16bf a1 = frag_ld(hidb, HDIM, mblk + 16, k, lane);
#pragma unroll
    for (int t = 0; t < 4; ++t) {
      v16bf b = frag_ld(WhTb, HDIM, nblk + 16 * t, k, lane);
      acc[0][t] = WMMA_BF16(a0, b, acc[0][t]);
      acc[1][t] = WMMA_BF16(a1, b, acc[1][t]);
    }
  }
#pragma unroll
  for (int mi = 0; mi < 2; ++mi)
#pragma unroll
    for (int t = 0; t < 4; ++t) {
      const int n = nblk + 16 * t + (lane & 15);
      const float bias = bh[n];
#pragma unroll
      for (int r = 0; r < 8; ++r) {
        const int m = mblk + mi * 16 + r + 8 * (lane >> 4);
        const float v = acc[mi][t][r] + bias;
        qkv[(long)m * (3 * HDIM) + n] = v;
        if (n < HDIM) qb16[(long)m * HDIM + n] = (__bf16)v;  // bf16 copy of q
      }
    }
}

// ---------------------------------------------------------------------------
// K2: qtil[p][t*256+e] = q[p] . W_t_key[e]   (8192x512 @ (768x512)^T)
// ---------------------------------------------------------------------------
__global__ __launch_bounds__(256) void k2_qtil(const __bf16* __restrict__ qb16,
                                               const __bf16* __restrict__ Wkb,
                                               float* __restrict__ qtil) {
  const int lane = threadIdx.x & 31, wid = threadIdx.x >> 5;
  const int mblk = blockIdx.y * 64 + (wid & 1) * 32;
  const int nblk = blockIdx.x * 256 + (wid >> 1) * 64;
  v8f acc[2][4] = {};
  for (int k = 0; k < HDIM; k += 32) {
    v16bf a0 = frag_ld(qb16, HDIM, mblk, k, lane);
    v16bf a1 = frag_ld(qb16, HDIM, mblk + 16, k, lane);
#pragma unroll
    for (int t = 0; t < 4; ++t) {
      v16bf b = frag_ld(Wkb, HDIM, nblk + 16 * t, k, lane);
      acc[0][t] = WMMA_BF16(a0, b, acc[0][t]);
      acc[1][t] = WMMA_BF16(a1, b, acc[1][t]);
    }
  }
#pragma unroll
  for (int mi = 0; mi < 2; ++mi)
#pragma unroll
    for (int t = 0; t < 4; ++t) {
      const int n = nblk + 16 * t + (lane & 15);
#pragma unroll
      for (int r = 0; r < 8; ++r) {
        const int m = mblk + mi * 16 + r + 8 * (lane >> 4);
        qtil[(long)m * 768 + n] = acc[mi][t][r];
      }
    }
}

// ---------------------------------------------------------------------------
// TDM gather: DMA up to 8 embedding rows (256 f32 each) into LDS.
// Descriptor per CDNA5 ISA (gather mode, 32-bit indices).
// ---------------------------------------------------------------------------
static __device__ __forceinline__ void tdm_gather_rows(const float* __restrict__ emb,
                                                       unsigned lds_off,
                                                       const int* tk, int nidx) {
  const unsigned long long ga = (unsigned long long)(uintptr_t)emb;
  u32x4 g0;
  g0[0] = 1u | (1u << 30) | (1u << 31);              // count=1, 32-bit idx, gather
  g0[1] = lds_off;                                   // lds_addr
  g0[2] = (unsigned)(ga & 0xffffffffu);              // global_addr[31:0]
  g0[3] = (unsigned)((ga >> 32) & 0x01ffffffu) | (2u << 30);  // addr[56:32], type=2
  i32x8 g1;
  g1[0] = (int)(2u << 16);                           // data_size = 4 bytes
  g1[1] = (int)(256u << 16);                         // tensor_dim0 = 256 (bits 79:48)
  g1[2] = (int)((VDIM & 0xffffu) << 16);             // tensor_dim1 lo (bits 95:80)
  g1[3] = (int)((VDIM >> 16) | (256u << 16));        // tensor_dim1 hi | tile_dim0=256
  g1[4] = nidx;                                      // tile_dim1 = #valid indices
  g1[5] = 256;                                       // tensor_dim0_stride
  g1[6] = 0;
  g1[7] = 0;
  i32x4 g2, g3;
  g2[0] = tk[0]; g2[1] = tk[1]; g2[2] = tk[2]; g2[3] = tk[3];
  g3[0] = tk[4]; g3[1] = tk[5]; g3[2] = tk[6]; g3[3] = tk[7];
#if __clang_major__ >= 23
  i32x8 gz = {};
  __builtin_amdgcn_tensor_load_to_lds(g0, g1, g2, g3, gz, 0);
#else
  __builtin_amdgcn_tensor_load_to_lds(g0, g1, g2, g3, 0);
#endif
}

// ---------------------------------------------------------------------------
// K3: per-position gates, scores, softmax, weighted-embedding accumulation.
// One block (8 waves) per position. Embedding rows DMA'd to LDS by the TDM
// while the block computes its 31 dot products, then reused twice from LDS.
// ---------------------------------------------------------------------------
static __device__ __forceinline__ float wave_reduce(float v) {
#pragma unroll
  for (int off = 16; off; off >>= 1) v += __shfl_xor(v, off, 32);
  return v;
}

__global__ __launch_bounds__(256) void k3_attn(
    const float* __restrict__ hiddens, const int* __restrict__ masks,
    const int* __restrict__ begins, const int* __restrict__ blens,
    const int* __restrict__ middles, const int* __restrict__ mlens,
    const int* __restrict__ ends, const int* __restrict__ elens,
    const float* __restrict__ emb, const float* __restrict__ Wg,
    const float* __restrict__ bg,
    const float* __restrict__ bb, const float* __restrict__ bm,
    const float* __restrict__ be,
    const float* __restrict__ qkv, const float* __restrict__ qtil,
    float* __restrict__ out, __bf16* __restrict__ accEb,
    float* __restrict__ wsum) {
  const int p = blockIdx.x;
  const int b = p >> 9;
  const int s = p & (SDIM - 1);
  const int tid = threadIdx.x, lane = tid & 31, wid = tid >> 5;

  __shared__ __align__(16) float esm[15 * EDIM];  // 15 gathered embedding rows
  __shared__ float qs[HDIM];
  __shared__ float dots[4];
  __shared__ float lmr[3][9];
  __shared__ float gates[3];
  __shared__ float sc[16];
  __shared__ float wts[16];
  __shared__ int   toks[16];

  if (tid < 15) {
    const int t = tid / 5, w = tid - t * 5;
    const int* tok = (t == 0) ? begins : (t == 1) ? middles : ends;
    toks[tid] = tok[(long)p * WDIM + w];
  } else if (tid == 15) {
    toks[15] = 0;  // padding index for the second (7-row) gather
  }
  const float* qrow = qkv + (long)p * (3 * HDIM);
  for (int i = tid; i < HDIM; i += 256) qs[i] = qrow[i];
  __syncthreads();

  // Kick off the embedding gather (2 TDM ops: rows 0-7 and 8-14).
  if (wid == 0) {
    const unsigned lds0 = (unsigned)(uintptr_t)&esm[0];  // LDS aperture: low 32 bits
    tdm_gather_rows(emb, lds0, &toks[0], 8);
    tdm_gather_rows(emb, lds0 + 8 * EDIM * 4, &toks[8], 7);
  }

  // Overlap: 4 q-dots (512) + 27 gate-logit dots (512) while the TDM runs.
  const int nlmr = (s >= 1 && s <= SDIM - 2) ? 27 : 0;
  for (int task = wid; task < 4 + nlmr; task += 8) {
    float ssum = 0.f;
    if (task < 4) {
      const float* v = (task == 0) ? (qrow + HDIM)
                     : (task == 1) ? bb : (task == 2) ? bm : be;  // key half
      for (int i = lane; i < HDIM; i += 32) ssum += qs[i] * v[i];
      ssum = wave_reduce(ssum);
      if (lane == 0) dots[task] = ssum;
    } else {
      const int j = task - 4, r = j / 9, c = j - r * 9;
      const float* hrow = hiddens + ((long)b * SDIM + (s - 1 + r)) * HDIM;
      for (int i = lane; i < HDIM; i += 32) ssum += hrow[i] * Wg[i * 9 + c];
      ssum = wave_reduce(ssum);
      if (lane == 0) lmr[r][c] = ssum + bg[c];
    }
  }
  if (wid == 0) __builtin_amdgcn_s_wait_tensorcnt(0);
  __syncthreads();

  if (tid == 0) {
    const float mf = (masks[p] == 0) ? 0.f : 1.f;  // pad positions -> gates zeroed
    float bgv, mgv, egv;
    if (s == 0 || s == SDIM - 1) { bgv = 1.f; mgv = 0.f; egv = 1.f; }
    else {
      bgv = 1.f / (1.f + __expf(-(lmr[0][0] + lmr[1][3] + lmr[2][6])));
      mgv = 1.f / (1.f + __expf(-(lmr[0][1] + lmr[1][4] + lmr[2][7])));
      egv = 1.f / (1.f + __expf(-(lmr[0][2] + lmr[1][5] + lmr[2][8])));
    }
    gates[0] = bgv * mf; gates[1] = mgv * mf; gates[2] = egv * mf;
    // hidden slot: score_mask true when masks != 0 (faithful to reference)
    sc[0] = (masks[p] != 0) ? -1e10f : dots[0];
  }
  __syncthreads();

  // Candidate scores from LDS: score = gate * (qtil . e + q.b_key)
  for (int j = wid; j < 15; j += 8) {
    const int t = j / 5, w = j - t * 5;
    const int* len = (t == 0) ? blens : (t == 1) ? mlens : elens;
    const int valid = (w < len[p]);  // reference masks VALID slots to -1e10 (faithful)
    const float* qt = qtil + (long)p * 768 + t * 256;
    float ssum = 0.f;
    for (int i = lane; i < EDIM; i += 32) ssum += qt[i] * esm[j * EDIM + i];
    ssum = wave_reduce(ssum);
    if (lane == 0) {
      const float scv = gates[t] * (ssum + dots[1 + t]);
      sc[1 + j] = valid ? -1e10f : scv;
    }
  }
  __syncthreads();

  if (tid == 0) {
    float mx = sc[0];
    for (int k = 1; k < 16; ++k) mx = fmaxf(mx, sc[k]);
    float sum = 0.f;
    for (int k = 0; k < 16; ++k) { wts[k] = __expf(sc[k] - mx); sum += wts[k]; }
    const float inv = 1.f / sum;
    for (int k = 0; k < 16; ++k) wts[k] *= inv;
  }
  __syncthreads();

  // out = w0 * hidden_value (value-GEMM contribution added by K4)
  const float* hv = qrow + 2 * HDIM;
  const float w0 = wts[0];
  for (int i = tid; i < HDIM; i += 256) out[(long)p * HDIM + i] = w0 * hv[i];

  // Weighted embedding sums per type (bf16 for K4's A operand), from LDS.
  for (int i = tid; i < 768; i += 256) {
    const int t = i >> 8, e = i & 255;
    float a = 0.f;
#pragma unroll
    for (int w = 0; w < WDIM; ++w)
      a += wts[1 + t * 5 + w] * esm[(t * 5 + w) * EDIM + e];
    accEb[(long)p * 768 + i] = (__bf16)(gates[t] * a);
  }
  if (tid < 3) {
    float wsf = 0.f;
#pragma unroll
    for (int w = 0; w < WDIM; ++w) wsf += wts[1 + tid * 5 + w];
    wsum[(long)p * 3 + tid] = gates[tid] * wsf;
  }
}

// ---------------------------------------------------------------------------
// K4: out += accE (8192x768) @ W_val_cat (768x512) + wsum_t * b_t_val
// ---------------------------------------------------------------------------
__global__ __launch_bounds__(256) void k4_value(const __bf16* __restrict__ accEb,
                                                const float* __restrict__ wsum,
                                                const __bf16* __restrict__ WvTb,
                                                const float* __restrict__ bb,
                                                const float* __restrict__ bm,
                                                const float* __restrict__ be,
                                                float* __restrict__ out) {
  const int lane = threadIdx.x & 31, wid = threadIdx.x >> 5;
  const int mblk = blockIdx.y * 64 + (wid & 1) * 32;
  const int nblk = blockIdx.x * 256 + (wid >> 1) * 64;
  v8f acc[2][4] = {};
  for (int k = 0; k < 768; k += 32) {
    v16bf a0 = frag_ld(accEb, 768, mblk, k, lane);
    v16bf a1 = frag_ld(accEb, 768, mblk + 16, k, lane);
#pragma unroll
    for (int t = 0; t < 4; ++t) {
      v16bf b = frag_ld(WvTb, 768, nblk + 16 * t, k, lane);
      acc[0][t] = WMMA_BF16(a0, b, acc[0][t]);
      acc[1][t] = WMMA_BF16(a1, b, acc[1][t]);
    }
  }
#pragma unroll
  for (int mi = 0; mi < 2; ++mi)
#pragma unroll
    for (int t = 0; t < 4; ++t) {
      const int n = nblk + 16 * t + (lane & 15);
      const float b0 = bb[HDIM + n], b1 = bm[HDIM + n], b2 = be[HDIM + n];
#pragma unroll
      for (int r = 0; r < 8; ++r) {
        const int m = mblk + mi * 16 + r + 8 * (lane >> 4);
        const float bias = wsum[(long)m * 3 + 0] * b0 +
                           wsum[(long)m * 3 + 1] * b1 +
                           wsum[(long)m * 3 + 2] * b2;
        out[(long)m * HDIM + n] += acc[mi][t][r] + bias;
      }
    }
}

// ---------------------------------------------------------------------------
extern "C" void kernel_launch(void* const* d_in, const int* in_sizes, int n_in,
                              void* d_out, int out_size, void* d_ws, size_t ws_size,
                              hipStream_t stream) {
  const float* hiddens = (const float*)d_in[0];
  const int*   masks   = (const int*)d_in[1];
  const int*   begins  = (const int*)d_in[2];
  const int*   blens   = (const int*)d_in[3];
  const int*   middles = (const int*)d_in[4];
  const int*   mlens   = (const int*)d_in[5];
  const int*   ends    = (const int*)d_in[6];
  const int*   elens   = (const int*)d_in[7];
  const float* emb     = (const float*)d_in[8];
  const float* Wh      = (const float*)d_in[9];
  const float* bh      = (const float*)d_in[10];
  const float* Wb      = (const float*)d_in[11];
  const float* bb      = (const float*)d_in[12];
  const float* Wm      = (const float*)d_in[13];
  const float* bm      = (const float*)d_in[14];
  const float* We      = (const float*)d_in[15];
  const float* be      = (const float*)d_in[16];
  const float* Wg      = (const float*)d_in[17];
  const float* bg      = (const float*)d_in[18];
  float* out = (float*)d_out;

  char* w = (char*)d_ws;
  float*  qkv   = (float*)w;  w += (size_t)BS * 1536 * 4;   // 50.3 MB
  float*  qtil  = (float*)w;  w += (size_t)BS * 768 * 4;    // 25.2 MB
  float*  wsum  = (float*)w;  w += (size_t)BS * 3 * 4;      //  0.1 MB
  __bf16* hidb  = (__bf16*)w; w += (size_t)BS * HDIM * 2;   //  8.4 MB
  __bf16* qb16  = (__bf16*)w; w += (size_t)BS * HDIM * 2;   //  8.4 MB
  __bf16* accEb = (__bf16*)w; w += (size_t)BS * 768 * 2;    // 12.6 MB
  __bf16* WhTb  = (__bf16*)w; w += (size_t)1536 * 512 * 2;  //  1.6 MB
  __bf16* Wkb   = (__bf16*)w; w += (size_t)768 * 512 * 2;   //  0.8 MB
  __bf16* WvTb  = (__bf16*)w; w += (size_t)512 * 768 * 2;   //  0.8 MB

  hipLaunchKernelGGL(k0_prep, dim3((K0_TOTAL + 255) / 256), dim3(256), 0, stream,
                     hiddens, Wh, Wb, Wm, We, hidb, WhTb, Wkb, WvTb);
  hipLaunchKernelGGL(k1_qkv, dim3(6, 128), dim3(256), 0, stream, hidb, WhTb, bh, qkv, qb16);
  hipLaunchKernelGGL(k2_qtil, dim3(3, 128), dim3(256), 0, stream, qb16, Wkb, qtil);
  hipLaunchKernelGGL(k3_attn, dim3(BS), dim3(256), 0, stream,
                     hiddens, masks, begins, blens, middles, mlens, ends, elens,
                     emb, Wg, bg, bb, bm, be, qkv, qtil, out, accEb, wsum);
  hipLaunchKernelGGL(k4_value, dim3(2, 128), dim3(256), 0, stream,
                     accEb, wsum, WvTb, bb, bm, be, out);
}
